// RationaleSelectorModel_64948495450288
// MI455X (gfx1250) — matrix operand order repaired and checked by reading, not compiled, not run on hardware
//
#include <hip/hip_runtime.h>
#include <hip/hip_bf16.h>
#include <math.h>

// Problem constants (match reference)
#define Bb 64
#define Tt 4096
#define Dd 256
#define Hh 341
#define Hp 352              // H padded to multiple of 16 (22 N-tiles)
#define NTILES (Hp / 16)    // 22
#define TAUc 1.0f
#define RHOc 0.3f
#define TVWc 0.1f
#define NEGc -1e9f
#define EPSc 1e-6f
#define LNEPSc 1e-5f

#define A_PITCH 264         // 256 + 8 bf16 pad -> conflict-reduced LDS A-tile

typedef __attribute__((ext_vector_type(16))) __bf16 v16bf;
typedef __attribute__((ext_vector_type(8)))  __bf16 v8bf;
typedef __attribute__((ext_vector_type(8)))  float  v8f;
typedef __attribute__((ext_vector_type(4)))  float  v4f;

// ---------------------------------------------------------------------------
// helpers
// ---------------------------------------------------------------------------
__device__ __forceinline__ float uhash01(unsigned x) {
    // deterministic hash -> uniform (0,1); stands in for jax PRNG(key=42)
    x ^= 42u * 0x9E3779B9u;
    x *= 0x85ebca6bu; x ^= x >> 13;
    x *= 0xc2b2ae35u; x ^= x >> 16;
    return ((float)(x >> 8) + 0.5f) * (1.0f / 16777216.0f);
}

__device__ __forceinline__ float blk_sum(float v, float* red, int tid) {
    red[tid] = v; __syncthreads();
    for (int off = 512; off > 0; off >>= 1) {
        if (tid < off) red[tid] += red[tid + off];
        __syncthreads();
    }
    float r = red[0]; __syncthreads();
    return r;
}

__device__ __forceinline__ float blk_max(float v, float* red, int tid) {
    red[tid] = v; __syncthreads();
    for (int off = 512; off > 0; off >>= 1) {
        if (tid < off) red[tid] = fmaxf(red[tid], red[tid + off]);
        __syncthreads();
    }
    float r = red[0]; __syncthreads();
    return r;
}

// descending bitonic sort of Tt floats in LDS, 1024 threads
__device__ __forceinline__ void bitonic_desc(float* a, int tid) {
    for (int k = 2; k <= Tt; k <<= 1) {
        for (int j = k >> 1; j > 0; j >>= 1) {
            #pragma unroll
            for (int s = 0; s < Tt / 1024; ++s) {
                int i = tid + s * 1024;
                int l = i ^ j;
                if (l > i) {
                    float ai = a[i], al = a[l];
                    bool desc = ((i & k) == 0);
                    bool sw = desc ? (ai < al) : (ai > al);
                    if (sw) { a[i] = al; a[l] = ai; }
                }
            }
            __syncthreads();
        }
    }
}

// ---------------------------------------------------------------------------
// Kernel 0: zero reg output (scores are written exactly once by the GEMM)
// ---------------------------------------------------------------------------
__global__ void rs_init_kernel(float* regOut) {
    if (blockIdx.x == 0 && threadIdx.x == 0) *regOut = 0.0f;
}

// ---------------------------------------------------------------------------
// Kernel 1: W1[D,H] f32 -> W1T[Hp,D] bf16 (transposed + zero padded)
// ---------------------------------------------------------------------------
__global__ void rs_w1t_kernel(const float* __restrict__ W1, __bf16* __restrict__ W1T) {
    int i = blockIdx.x * blockDim.x + threadIdx.x;
    if (i >= Hp * Dd) return;
    int n = i / Dd, k = i % Dd;
    float v = (n < Hh) ? W1[k * Hh + n] : 0.0f;
    W1T[(size_t)n * Dd + k] = (__bf16)v;
}

// ---------------------------------------------------------------------------
// Kernel 2: FUSED  mask + layernorm -> LDS bf16 A-tile -> WMMA GEMM over all
//           22 N-tiles -> GELU -> W2-dot -> scores[16 rows], one block/M-tile.
//   256 threads = 8 waves. LN: 2 rows per wave, wave-level shfl reductions.
//   GEMM: wave w sweeps N-tiles {w, w+8, w+16}; A from LDS, B from L2-hot W1T.
//   Scores accumulated in LDS (ds_add_f32), stored once. No xn round-trip.
// ---------------------------------------------------------------------------
__global__ void __launch_bounds__(256) rs_fused_kernel(
        const float* __restrict__ emb, const float* __restrict__ attn,
        const float* __restrict__ gamma, const float* __restrict__ beta,
        const __bf16* __restrict__ W1T, const float* __restrict__ b1,
        const float* __restrict__ W2, float* __restrict__ scoresAcc) {
    __shared__ __align__(16) __bf16 aTile[16][A_PITCH];  // 16 x 256 (+pad) bf16
    __shared__ float sscore[16];

    const int tid  = threadIdx.x;
    const int wave = tid >> 5;
    const int lane = tid & 31;
    const int m    = blockIdx.x;                 // M-tile: rows m*16 .. m*16+15

    if (tid < 16) sscore[tid] = 0.0f;

    // ---- Phase 1: layernorm 2 rows per wave, all reductions intra-wave ----
    #pragma unroll
    for (int rr = 0; rr < 2; ++rr) {
        int r    = wave * 2 + rr;                // row within tile, 0..15
        int grow = m * 16 + r;                   // global row
        float a  = attn[grow];
        const float* ep = emb + (size_t)grow * Dd + lane * 8;
        v4f e0 = *(const v4f*)(ep);
        v4f e1 = *(const v4f*)(ep + 4);
        float e[8];
        #pragma unroll
        for (int i = 0; i < 4; ++i) { e[i] = e0[i] * a; e[4 + i] = e1[i] * a; }
        float s = 0.0f, sq = 0.0f;
        #pragma unroll
        for (int i = 0; i < 8; ++i) { s += e[i]; sq += e[i] * e[i]; }
        #pragma unroll
        for (int off = 1; off < 32; off <<= 1) {
            s  += __shfl_xor(s,  off, 32);
            sq += __shfl_xor(sq, off, 32);
        }
        float mu   = s * (1.0f / Dd);
        float var  = sq * (1.0f / Dd) - mu * mu;
        float rstd = rsqrtf(var + LNEPSc);
        const float* gp = gamma + lane * 8;
        const float* bp = beta + lane * 8;
        v4f g0 = *(const v4f*)(gp),     g1 = *(const v4f*)(gp + 4);
        v4f p0 = *(const v4f*)(bp),     p1 = *(const v4f*)(bp + 4);
        v8bf xv;
        #pragma unroll
        for (int i = 0; i < 4; ++i) {
            xv[i]     = (__bf16)((e[i]     - mu) * rstd * g0[i] + p0[i]);
            xv[4 + i] = (__bf16)((e[4 + i] - mu) * rstd * g1[i] + p1[i]);
        }
        *(v8bf*)(&aTile[r][lane * 8]) = xv;
    }
    __syncthreads();

    // ---- Phase 2: each wave sweeps its N-tiles against the LDS A-tile ----
    const int half = lane >> 4;                  // 0: lanes 0-15, 1: 16-31
    const int l16  = lane & 15;
    const __bf16* arow = &aTile[l16][0];         // A row (M = l16) in LDS

    for (int nt = wave; nt < NTILES; nt += 8) {
        const __bf16* brow = W1T + ((size_t)(nt * 16 + l16)) * Dd;  // B col (N)
        v8f acc = {};
        #pragma unroll
        for (int k0 = 0; k0 < Dd; k0 += 32) {
            // A 16x32 bf16 fragment: half 0 -> K {0..7,16..23}, half 1 -> {8..15,24..31}
            v8bf a0 = *(const v8bf*)(arow + k0 + half * 8);
            v8bf a1 = *(const v8bf*)(arow + k0 + 16 + half * 8);
            // B 32x16 bf16 fragment: half 0 -> K {0..7,16..23}, half 1 -> {8..15,24..31}
            v8bf b0 = *(const v8bf*)(brow + k0 + half * 8);
            v8bf b1v = *(const v8bf*)(brow + k0 + 16 + half * 8);
            v16bf av, bv;
            #pragma unroll
            for (int i = 0; i < 8; ++i) {
                av[i] = a0[i]; av[8 + i] = a1[i];
                bv[i] = b0[i]; bv[8 + i] = b1v[i];
            }
            acc = __builtin_amdgcn_wmma_f32_16x16x32_bf16(
                      false, av, false, bv, (short)0, acc, false, false);
        }

        // epilogue: h = gelu(acc + b1[n]); partial score += h * W2[n]
        int ncol   = nt * 16 + l16;
        bool valid = (ncol < Hh);
        float w2v  = valid ? W2[ncol] : 0.0f;
        float b1v  = valid ? b1[ncol] : 0.0f;
        float contrib[8];
        #pragma unroll
        for (int j = 0; j < 8; ++j) {
            float v = acc[j] + b1v;
            float g = 0.5f * v * (1.0f + erff(v * 0.70710678118f));  // exact gelu
            contrib[j] = valid ? g * w2v : 0.0f;
        }
        // butterfly reduce across the 16 N-lanes of each half-wave
        #pragma unroll
        for (int off = 1; off < 16; off <<= 1) {
            #pragma unroll
            for (int j = 0; j < 8; ++j)
                contrib[j] += __shfl_xor(contrib[j], off, 16);
        }
        if (l16 == 0) {
            #pragma unroll
            for (int j = 0; j < 8; ++j)
                atomicAdd(&sscore[half * 8 + j], contrib[j]);   // ds_add_f32
        }
    }
    __syncthreads();
    if (tid < 16) scoresAcc[(size_t)m * 16 + tid] = sscore[tid];
}

// ---------------------------------------------------------------------------
// Kernel 3: per batch row: entmax-1.5 (sort-based), gumbel var-k top-k, TV reg
//   one workgroup (1024 thr = 32 waves) per row; sorts/scans in LDS.
// ---------------------------------------------------------------------------
__global__ void __launch_bounds__(1024) rs_entmax_kernel(
        const float* __restrict__ scoresAcc, const float* __restrict__ attn,
        const float* __restrict__ b2p,
        float* __restrict__ zOut, float* __restrict__ gOut, float* regOut) {
    __shared__ float  xs[Tt];        // sort buffer
    __shared__ float  c1[Tt];        // cumsum / g buffer
    __shared__ float  c2[Tt];        // cumsum-sq / tau buffer
    __shared__ float2 pp[1024];      // scan partials
    __shared__ float  red[1024];     // reductions
    __shared__ float  shv[2];

    const int NTHR = 1024;
    int b = blockIdx.x;
    int tid = threadIdx.x;
    float b2 = b2p[0];
    const float* sc = scoresAcc + (size_t)b * Tt;
    const float* am = attn + (size_t)b * Tt;

    // t_eff and row max of s/TAU
    float te = 0.0f, mx = -3.4e38f;
    for (int t = tid; t < Tt; t += NTHR) {
        float a = am[t];
        te += a;
        float s = (a != 0.0f) ? (sc[t] + b2) : NEGc;
        mx = fmaxf(mx, s * (1.0f / TAUc));
    }
    te = blk_sum(te, red, tid);
    mx = blk_max(mx, red, tid);
    if (tid == 0) { shv[0] = mx; shv[1] = te; }
    __syncthreads();
    mx = shv[0]; te = shv[1];

    // x = (s/TAU - max)/2 into xs, then descending sort
    for (int t = tid; t < Tt; t += NTHR) {
        float a = am[t];
        float s = (a != 0.0f) ? (sc[t] + b2) : NEGc;
        xs[t] = (s * (1.0f / TAUc) - mx) * 0.5f;
    }
    __syncthreads();
    bitonic_desc(xs, tid);

    // inclusive scans of xs and xs^2 (two-level: 4 elems/thread + H-S scan)
    {
        int base = tid * 4;
        float s1 = 0.0f, s2 = 0.0f, l1[4], l2[4];
        #pragma unroll
        for (int i = 0; i < 4; ++i) {
            float v = xs[base + i];
            s1 += v;     l1[i] = s1;
            s2 += v * v; l2[i] = s2;
        }
        pp[tid] = make_float2(s1, s2);
        __syncthreads();
        for (int off = 1; off < 1024; off <<= 1) {
            float ax = 0.0f, ay = 0.0f;
            if (tid >= off) { float2 p = pp[tid - off]; ax = p.x; ay = p.y; }
            __syncthreads();
            if (tid >= off) { pp[tid].x += ax; pp[tid].y += ay; }
            __syncthreads();
        }
        float ox = 0.0f, oy = 0.0f;
        if (tid > 0) { float2 p = pp[tid - 1]; ox = p.x; oy = p.y; }
        #pragma unroll
        for (int i = 0; i < 4; ++i) {
            c1[base + i] = ox + l1[i];
            c2[base + i] = oy + l2[i];
        }
    }
    __syncthreads();

    // tau per prefix (overwrite c2), support count, tau*
    {
        int base = tid * 4;
        #pragma unroll
        for (int i = 0; i < 4; ++i) {
            int idx = base + i;
            float rho  = (float)(idx + 1);
            float mean = c1[idx] / rho;
            float msq  = c2[idx] / rho;
            float ss   = rho * (msq - mean * mean);
            float delta = (1.0f - ss) / rho;
            c2[idx] = mean - sqrtf(fmaxf(delta, 0.0f));
        }
    }
    __syncthreads();
    float cnt = 0.0f;
    for (int t = tid; t < Tt; t += NTHR)
        cnt += (c2[t] <= xs[t]) ? 1.0f : 0.0f;
    cnt = blk_sum(cnt, red, tid);
    int support = (int)cnt; if (support < 1) support = 1;
    float tau_star = c2[support - 1];

    // z = max(x - tau*, 0)^2 * attn
    for (int t = tid; t < Tt; t += NTHR) {
        float a = am[t];
        float s = (a != 0.0f) ? (sc[t] + b2) : NEGc;
        float x = (s * (1.0f / TAUc) - mx) * 0.5f;
        float d = fmaxf(x - tau_star, 0.0f);
        zOut[(size_t)b * Tt + t] = d * d * a;
    }
    __syncthreads();

    // gumbel-perturbed variable-k top-k
    float kr = fminf(fmaxf(roundf(RHOc * te), 1.0f), fmaxf(te, 1.0f));
    int kk = (int)kr;
    for (int t = tid; t < Tt; t += NTHR) {
        float a = am[t];
        float s = (a != 0.0f) ? (sc[t] + b2) : NEGc;
        float u = uhash01((unsigned)(b * Tt + t));
        float gum = -logf(-logf(u + EPSc) + EPSc);
        xs[t] = s * a + gum;
    }
    __syncthreads();
    bitonic_desc(xs, tid);
    float thr = xs[kk - 1];

    // h (== forward value of g); keep in c1 for TV
    for (int t = tid; t < Tt; t += NTHR) {
        float a = am[t];
        float s = (a != 0.0f) ? (sc[t] + b2) : NEGc;
        float u = uhash01((unsigned)(b * Tt + t));
        float gum = -logf(-logf(u + EPSc) + EPSc);
        float pert = s * a + gum;
        float h = (pert >= thr) ? a : 0.0f;
        c1[t] = h;
        gOut[(size_t)b * Tt + t] = h;
    }
    __syncthreads();

    // total-variation regularizer for this row
    float s1 = 0.0f, s2 = 0.0f;
    for (int t = tid + 1; t < Tt; t += NTHR) {
        float valid = am[t] * am[t - 1];
        s1 += fabsf(c1[t] - c1[t - 1]) * valid;
        s2 += valid;
    }
    s1 = blk_sum(s1, red, tid);
    s2 = blk_sum(s2, red, tid);
    if (tid == 0) {
        float tv = s1 / fmaxf(s2, 1.0f);
        atomicAdd(regOut, TVWc * tv * (1.0f / Bb));
    }
}

// ---------------------------------------------------------------------------
// launcher
// ---------------------------------------------------------------------------
extern "C" void kernel_launch(void* const* d_in, const int* in_sizes, int n_in,
                              void* d_out, int out_size, void* d_ws, size_t ws_size,
                              hipStream_t stream) {
    const float* emb   = (const float*)d_in[0];   // [B,T,D]
    const float* attn  = (const float*)d_in[1];   // [B,T]
    const float* gamma = (const float*)d_in[2];   // [D]
    const float* beta  = (const float*)d_in[3];   // [D]
    const float* W1    = (const float*)d_in[4];   // [D,H]
    const float* b1    = (const float*)d_in[5];   // [H]
    const float* W2    = (const float*)d_in[6];   // [H]
    const float* b2    = (const float*)d_in[7];   // scalar

    float* out    = (float*)d_out;
    float* zOut   = out;                          // [B,T]
    float* gOut   = out + (size_t)Bb * Tt;        // [B,T]
    float* regOut = out + (size_t)2 * Bb * Tt;    // scalar

    // workspace layout (no xn intermediate anymore)
    char* ws = (char*)d_ws;
    __bf16* W1T = (__bf16*)ws;                                    // 176 KB
    size_t off = (size_t)Hp * Dd * sizeof(__bf16);
    off = (off + 255) & ~(size_t)255;
    float* scoresAcc = (float*)(ws + off);                        // 1 MB

    const int nrows = Bb * Tt;

    rs_init_kernel<<<1, 64, 0, stream>>>(regOut);
    rs_w1t_kernel<<<(Hp * Dd + 255) / 256, 256, 0, stream>>>(W1, W1T);
    rs_fused_kernel<<<nrows / 16, 256, 0, stream>>>(
        emb, attn, gamma, beta, W1T, b1, W2, scoresAcc);
    rs_entmax_kernel<<<Bb, 1024, 0, stream>>>(scoresAcc, attn, b2, zOut, gOut, regOut);
}
